// SingleAttention_19207093747888
// MI455X (gfx1250) — matrix operand order repaired and checked by reading, not compiled, not run on hardware
//
#include <hip/hip_runtime.h>
#include <hip/hip_bf16.h>
#include <math.h>

typedef _Float16 half_t;
typedef __attribute__((ext_vector_type(16))) _Float16 v16h;
typedef __attribute__((ext_vector_type(8)))  _Float16 v8h;
typedef __attribute__((ext_vector_type(8)))  float    v8f;
typedef __attribute__((ext_vector_type(4)))  unsigned int v4u;
typedef __attribute__((ext_vector_type(8)))  int v8i;
typedef __attribute__((ext_vector_type(4)))  int v4i;

#define C_DIM 256
#define HW    1600
#define SCALE 0.0625f   // 256^-0.5

// ---- WMMA fragment helpers (per CDNA5 ISA 7.12.2 layouts, wave32) ----
__device__ __forceinline__ v16h load_afrag(const half_t* row, int k0, int hi) {
    union { v16h v; v8h h[2]; } u;
    u.h[0] = *(const v8h*)(row + k0 + hi * 8);
    u.h[1] = *(const v8h*)(row + k0 + 16 + hi * 8);
    return u.v;
}
__device__ __forceinline__ v16h load_bfrag(const half_t* btrow, int k0, int hi) {
    union { v16h v; v8h h[2]; } u;
    u.h[0] = *(const v8h*)(btrow + k0 + hi * 8);
    u.h[1] = *(const v8h*)(btrow + k0 + 16 + hi * 8);
    return u.v;
}
__device__ __forceinline__ v8f wmma32(v16h a, v16h b, v8f c) {
    return __builtin_amdgcn_wmma_f32_16x16x32_f16(false, a, false, b, (short)0, c,
                                                  false, false);
}

// ---- Kernel 1: weight f32->f16 + sincos positional embedding table ----
__global__ void prep_kernel(const float* __restrict__ qkv_w,
                            half_t* __restrict__ wh, float* __restrict__ pe) {
    int idx = blockIdx.x * blockDim.x + threadIdx.x;
    if (idx < 3 * C_DIM * C_DIM) wh[idx] = (half_t)qkv_w[idx];
    if (idx < HW * C_DIM) {
        int c = idx & (C_DIM - 1);
        int t = idx >> 8;             // t = h*40 + w
        int hh = t / 40, ww = t % 40;
        int grp = c >> 6, cm = c & 63;
        float om  = __powf(10000.f, -(float)cm * (1.0f / 64.0f));
        float arg = (grp < 2 ? (float)ww : (float)hh) * om;
        pe[idx] = ((grp & 1) == 0) ? __sinf(arg) : __cosf(arg);
    }
}

// ---- Kernel 2: LayerNorm + QKV GEMM (64 pixels per block, 4 waves) ----
__global__ void __launch_bounds__(128) lnqkv_kernel(
    const float* __restrict__ x, const float* __restrict__ ln_w,
    const float* __restrict__ ln_b, const half_t* __restrict__ wh,
    const float* __restrict__ qkv_b, const float* __restrict__ pe,
    half_t* __restrict__ qh, half_t* __restrict__ kh, half_t* __restrict__ vT) {
    __shared__ half_t xn[64 * 256];
    __shared__ float psum[128], psum2[128];
    __shared__ float meanL[64], rstdL[64];

    const int tid   = threadIdx.x;
    const int pbase = blockIdx.x * 64;   // 1600 % 64 == 0 -> tile inside one batch
    const int bb    = pbase / HW;
    const int hw0   = pbase % HW;
    const size_t xbase = (size_t)bb * C_DIM * HW;

    // channel stats: 2 threads per pixel
    {
        int pl = tid >> 1, hf = tid & 1;
        size_t off = xbase + (size_t)(hw0 + pl);
        float s = 0.f, s2 = 0.f;
        for (int c = hf * 128; c < hf * 128 + 128; ++c) {
            float v = x[off + (size_t)c * HW];
            s += v; s2 += v * v;
        }
        psum[tid] = s; psum2[tid] = s2;
    }
    __syncthreads();
    if ((tid & 1) == 0) {
        int pl = tid >> 1;
        float s  = psum[tid] + psum[tid + 1];
        float s2 = psum2[tid] + psum2[tid + 1];
        float mu = s * (1.f / 256.f);
        float var = s2 * (1.f / 256.f) - mu * mu;
        meanL[pl] = mu;
        rstdL[pl] = rsqrtf(var + 1e-5f);
    }
    __syncthreads();
    for (int idx = tid; idx < 64 * 256; idx += 128) {
        int pl = idx & 63;
        int c  = idx >> 6;
        float v  = x[xbase + (size_t)c * HW + hw0 + pl];
        float nv = (v - meanL[pl]) * rstdL[pl] * ln_w[c] + ln_b[c];
        xn[pl * 256 + c] = (half_t)nv;
    }
    __syncthreads();

    const int lane = tid & 31;
    const int wv   = tid >> 5;
    const int hi   = lane >> 4;
    const int nl   = lane & 15;
    const int m0   = wv * 16;

    // hoist A fragments (16 rows x K=256) into registers
    const half_t* arow = xn + (m0 + nl) * 256;
    v16h afr[8];
#pragma unroll
    for (int kt = 0; kt < 8; ++kt) afr[kt] = load_afrag(arow, kt * 32, hi);

    auto epilogue = [&](int nt, const v8f& acc) {
        const int which = nt >> 4;           // wave-uniform: 0=q, 1=k, 2=v
        const int o     = nt * 16 + nl;
        const int co    = o & 255;
        const float bias = qkv_b[o];
#pragma unroll
        for (int r = 0; r < 8; ++r) {
            int m = m0 + r + hi * 8;
            int p = pbase + m;
            int t = hw0 + m;
            float val = acc[r] + bias;
            if (which == 0) {
                val = (val + pe[t * 256 + co]) * SCALE;
                qh[(size_t)p * 256 + co] = (half_t)val;
            } else if (which == 1) {
                val += pe[t * 256 + co];
                kh[(size_t)p * 256 + co] = (half_t)val;
            } else {
                vT[((size_t)bb * 256 + co) * HW + t] = (half_t)val;
            }
        }
    };

    // 48 output tiles, processed in pairs with batched B-fragment loads
    for (int ntp = 0; ntp < 24; ++ntp) {
        const int nt0 = ntp * 2;
        v8f acc0 = {}, acc1 = {};
        const half_t* bt0 = wh + (size_t)(nt0 * 16 + nl) * 256;
        const half_t* bt1 = bt0 + 16 * 256;
#pragma unroll
        for (int g = 0; g < 2; ++g) {
            v16h b0[4], b1[4];
#pragma unroll
            for (int i = 0; i < 4; ++i) {
                b0[i] = load_bfrag(bt0, (g * 4 + i) * 32, hi);
                b1[i] = load_bfrag(bt1, (g * 4 + i) * 32, hi);
            }
#pragma unroll
            for (int i = 0; i < 4; ++i) {
                acc0 = wmma32(afr[g * 4 + i], b0[i], acc0);
                acc1 = wmma32(afr[g * 4 + i], b1[i], acc1);
            }
        }
        epilogue(nt0, acc0);
        epilogue(nt0 + 1, acc1);
    }
}

// ---- Kernel 3: attention, 64 queries/block, 8 waves (2 halves of keyspace) ----
__global__ void __launch_bounds__(256) attn_kernel(
    const half_t* __restrict__ qh, const half_t* __restrict__ kh,
    const half_t* __restrict__ vT, const float* __restrict__ x,
    const float* __restrict__ gamma, float* __restrict__ out) {
    extern __shared__ char smem[];
    half_t* qt = (half_t*)smem;                         // 64*256 f16 = 32 KB
    half_t* P  = (half_t*)(smem + 32768);               // 64*1600 f16 = 200 KB
    float* rmaxbuf = (float*)(smem + 32768 + 204800);   // [2][64]
    float* rowsum  = rmaxbuf + 128;                     // [64]

    const int tid  = threadIdx.x;
    const int lane = tid & 31;
    const int wv   = tid >> 5;      // 0..7
    const int hi   = lane >> 4;
    const int nl   = lane & 15;
    const int rw   = wv & 3;        // row group (16 rows)
    const int kv   = wv >> 2;       // key/col half
    const int m0   = rw * 16;
    const int t0   = blockIdx.x * 64;
    const int b    = blockIdx.y;

    // ---- Q tile: 32KB contiguous global->LDS via Tensor Data Mover ----
    if (wv == 0) {
        const unsigned long long ga =
            (unsigned long long)(uintptr_t)(qh + ((size_t)b * HW + t0) * 256);
        const unsigned int lds_off = (unsigned int)(uintptr_t)qt;  // LDS byte offset
        v4u g0;
        g0[0] = 1u;                                   // count=1, user descriptor
        g0[1] = lds_off;                              // lds_addr
        g0[2] = (unsigned int)ga;                     // global_addr[31:0]
        g0[3] = (unsigned int)((ga >> 32) & 0x1FFFFFFu) | (2u << 30); // addr|type=2
        v8i g1;
        g1[0] = 0x00010000;                 // data_size=1 (2B); mask=0; no flags
        g1[1] = (int)(16384u << 16);        // tensor_dim0[15:0]=16384 (elements)
        g1[2] = 0x00010000;                 // tensor_dim0[31:16]=0 | tensor_dim1=1
        g1[3] = (int)(16384u << 16);        // tile_dim0 = 16384
        g1[4] = 1;                          // tile_dim1 = 1
        g1[5] = 16384;                      // tensor_dim0_stride[31:0]
        g1[6] = 0;
        g1[7] = 0;
        v4i gz = {0, 0, 0, 0};
#if __clang_major__ >= 23
        v8i gz8 = {0, 0, 0, 0, 0, 0, 0, 0};
        __builtin_amdgcn_tensor_load_to_lds(g0, g1, gz, gz, gz8, 0);
#else
        __builtin_amdgcn_tensor_load_to_lds(g0, g1, gz, gz, 0);
#endif
        __builtin_amdgcn_s_wait_tensorcnt(0);
    }
    __syncthreads();

    // hoist Q fragments into registers
    const half_t* arow = qt + (m0 + nl) * 256;
    v16h afr[8];
#pragma unroll
    for (int kt = 0; kt < 8; ++kt) afr[kt] = load_afrag(arow, kt * 32, hi);

    // S = Q*K^T : this wave covers keys [kv*800, kv*800+800), 50 tiles, in pairs
    float rmax[8];
#pragma unroll
    for (int r = 0; r < 8; ++r) rmax[r] = -1e30f;
    for (int ntp = 0; ntp < 25; ++ntp) {
        const int nt0 = kv * 50 + ntp * 2;
        v8f acc0 = {}, acc1 = {};
        const half_t* bt0 = kh + ((size_t)b * HW + nt0 * 16 + nl) * 256;
        const half_t* bt1 = bt0 + 16 * 256;
#pragma unroll
        for (int g = 0; g < 2; ++g) {
            v16h b0[4], b1[4];
#pragma unroll
            for (int i = 0; i < 4; ++i) {
                b0[i] = load_bfrag(bt0, (g * 4 + i) * 32, hi);
                b1[i] = load_bfrag(bt1, (g * 4 + i) * 32, hi);
            }
#pragma unroll
            for (int i = 0; i < 4; ++i) {
                acc0 = wmma32(afr[g * 4 + i], b0[i], acc0);
                acc1 = wmma32(afr[g * 4 + i], b1[i], acc1);
            }
        }
#pragma unroll
        for (int r = 0; r < 8; ++r) {
            int m = m0 + r + hi * 8;
            P[m * 1600 + nt0 * 16 + nl]        = (half_t)acc0[r];
            P[m * 1600 + (nt0 + 1) * 16 + nl]  = (half_t)acc1[r];
            rmax[r] = fmaxf(rmax[r], fmaxf(acc0[r], acc1[r]));
        }
    }
    // reduce per-row max over the 16 lanes of each half-wave; 2 slots (key halves)
#pragma unroll
    for (int r = 0; r < 8; ++r) {
        float v = rmax[r];
        for (int off = 8; off >= 1; off >>= 1)
            v = fmaxf(v, __shfl_xor(v, off, 16));
        if (nl == 0) rmaxbuf[kv * 64 + m0 + r + hi * 8] = v;
    }
    __syncthreads();

    // softmax numerator + row sums: 8 waves x 8 rows each, packed-pair LDS ops
    {
        unsigned int* P32 = (unsigned int*)P;
        for (int rr = 0; rr < 8; ++rr) {
            int m = wv * 8 + rr;
            float rm = fmaxf(rmaxbuf[m], rmaxbuf[64 + m]);
            float s = 0.f;
            for (int j = lane; j < 800; j += 32) {
                union { unsigned int u; _Float16 h[2]; } pk;
                pk.u = P32[m * 800 + j];
                float e0 = __expf((float)pk.h[0] - rm);
                float e1 = __expf((float)pk.h[1] - rm);
                pk.h[0] = (half_t)e0;
                pk.h[1] = (half_t)e1;
                P32[m * 800 + j] = pk.u;
                s += e0 + e1;
            }
            for (int off = 16; off >= 1; off >>= 1) s += __shfl_xor(s, off, 32);
            if (lane == 0) rowsum[m] = s;
        }
    }
    __syncthreads();

    // O = P*V : wave covers col tiles [kv*8, kv*8+8), rows m0..m0+15
    const float g = gamma[0];
    const half_t* prow = P + (m0 + nl) * 1600;
    for (int ntp = 0; ntp < 4; ++ntp) {
        const int nt0 = kv * 8 + ntp * 2;
        v8f acc0 = {}, acc1 = {};
        const half_t* bt0 = vT + ((size_t)b * 256 + nt0 * 16 + nl) * HW;
        const half_t* bt1 = bt0 + 16 * HW;
        for (int kt = 0; kt < 50; ++kt) {
            v16h a  = load_afrag(prow, kt * 32, hi);
            v16h f0 = load_bfrag(bt0, kt * 32, hi);
            v16h f1 = load_bfrag(bt1, kt * 32, hi);
            acc0 = wmma32(a, f0, acc0);
            acc1 = wmma32(a, f1, acc1);
        }
#pragma unroll
        for (int r = 0; r < 8; ++r) {
            int m  = m0 + r + hi * 8;
            int t  = t0 + m;
            float inv = 1.0f / rowsum[m];
            int cc0 = nt0 * 16 + nl;
            size_t idx0 = ((size_t)b * 256 + cc0) * HW + t;
            size_t idx1 = idx0 + (size_t)16 * HW;
            out[idx0] = x[idx0] + g * (acc0[r] * inv);
            out[idx1] = x[idx1] + g * (acc1[r] * inv);
        }
    }
}

extern "C" void kernel_launch(void* const* d_in, const int* in_sizes, int n_in,
                              void* d_out, int out_size, void* d_ws, size_t ws_size,
                              hipStream_t stream) {
    const float* x     = (const float*)d_in[0];
    const float* ln_w  = (const float*)d_in[1];
    const float* ln_b  = (const float*)d_in[2];
    const float* qkv_w = (const float*)d_in[3];
    const float* qkv_b = (const float*)d_in[4];
    const float* gamma = (const float*)d_in[5];
    float* out = (float*)d_out;

    char* ws = (char*)d_ws;
    half_t* wh = (half_t*)ws;                          // 768*256*2   =   393216
    float*  pe = (float*)(ws + 393216);                // 1600*256*4  =  1638400
    half_t* qh = (half_t*)(ws + 2031616);              // 25600*256*2 = 13107200
    half_t* kh = (half_t*)(ws + 15138816);             // 13107200
    half_t* vT = (half_t*)(ws + 28246016);             // 13107200  (total ~41.4MB)

    prep_kernel<<<1600, 256, 0, stream>>>(qkv_w, wh, pe);
    lnqkv_kernel<<<400, 128, 0, stream>>>(x, ln_w, ln_b, wh, qkv_b, pe, qh, kh, vT);
    attn_kernel<<<dim3(25, 16), 256, 238336, stream>>>(qh, kh, vT, x, gamma, out);
}
